// ResShortCut_AttenSpconv_Dec_73701638799742
// MI455X (gfx1250) — matrix operand order repaired and checked by reading, not compiled, not run on hardware
//
#include <hip/hip_runtime.h>

typedef _Float16 v16h __attribute__((ext_vector_type(16)));
typedef _Float16 v8h  __attribute__((ext_vector_type(8)));
typedef float    v8f  __attribute__((ext_vector_type(8)));
typedef unsigned int u32x4 __attribute__((ext_vector_type(4)));
typedef int          i32x4 __attribute__((ext_vector_type(4)));
typedef int          i32x8 __attribute__((ext_vector_type(8)));

#define BN_S 0.9999950000374997f  /* 1/sqrt(1+1e-5) */

enum { EP_RELU_BN = 0, EP_LEAKY_BN = 1, EP_ADD = 2, EP_WHERE = 3 };

#if defined(__HIP_DEVICE_COMPILE__) && __has_builtin(__builtin_amdgcn_tensor_load_to_lds) && \
    __has_builtin(__builtin_amdgcn_s_wait_tensorcnt)
#define ATHENA_TDM 1
#else
#define ATHENA_TDM 0
#endif

__device__ inline v8h zero_v8h() {
    v8h z;
#pragma unroll
    for (int e = 0; e < 8; ++e) z[e] = (_Float16)0.0f;
    return z;
}

#if ATHENA_TDM
// Issue one TDM 1-row 2D tile load: dw dwords from gsrc (global) to ldst (LDS).
// D# per CDNA5 ISA 8.3-8.5: group0 = {count=1, lds_addr, global_addr, type=2},
// group1 = {data_size=4B, tensor_dim0=tile_dim0=dw, tile_dim1=1}.
__device__ inline void tdm_load_row(const void* gsrc, void* ldst, unsigned dw) {
    unsigned lds = (unsigned)(unsigned long long)(size_t)ldst;  // low 32b = LDS offset
    unsigned long long ga = (unsigned long long)(size_t)gsrc;
    u32x4 g0;
    g0[0] = 1u;                                                // count=1 (user mode)
    g0[1] = lds;                                               // lds_addr (bytes)
    g0[2] = (unsigned)(ga & 0xFFFFFFFFu);                      // global_addr[31:0]
    g0[3] = (unsigned)((ga >> 32) & 0x1FFFFFFu) | (2u << 30);  // addr[56:32], type=2
    i32x8 g1;
    g1[0] = (2 << 16);                                         // data_size = 4 bytes
    g1[1] = (int)((dw & 0xFFFFu) << 16);                       // tensor_dim0[15:0]
    g1[2] = (int)((dw >> 16) & 0xFFFFu) | (1 << 16);           // td0[31:16], tensor_dim1=1
    g1[3] = (int)((dw & 0xFFFFu) << 16);                       // tile_dim0 = dw
    g1[4] = 1;                                                 // tile_dim1 = 1, tile_dim2 = 0
    g1[5] = (int)dw;                                           // tensor_dim0_stride (unused)
    g1[6] = 0;
    g1[7] = 0;
    i32x4 z4; z4[0] = z4[1] = z4[2] = z4[3] = 0;
#if defined(__clang_major__) && (__clang_major__ >= 23)
    i32x8 z8;
#pragma unroll
    for (int e = 0; e < 8; ++e) z8[e] = 0;
    __builtin_amdgcn_tensor_load_to_lds(g0, g1, z4, z4, z8, 0);
#else
    __builtin_amdgcn_tensor_load_to_lds(g0, g1, z4, z4, 0);
#endif
}
#endif

// ---------------------------------------------------------------------------
// mask1 = (roi > 0.8) ? 1 : 0   (f16)
// ---------------------------------------------------------------------------
__global__ __launch_bounds__(256) void build_mask1_k(const float* __restrict__ roi,
                                                     _Float16* __restrict__ m1, int total) {
    int idx = blockIdx.x * 256 + threadIdx.x;
    if (idx >= total) return;
    m1[idx] = (_Float16)((roi[idx] > 0.8f) ? 1.0f : 0.0f);
}

// ---------------------------------------------------------------------------
// 3x3 / stride-2 / pad-1 max pool (active-site generation rule)
// ---------------------------------------------------------------------------
__global__ __launch_bounds__(256) void pool_mask_k(const _Float16* __restrict__ in,
                                                   _Float16* __restrict__ out,
                                                   int N, int Hin, int Win, int Hout, int Wout) {
    int idx = blockIdx.x * 256 + threadIdx.x;
    int total = N * Hout * Wout;
    if (idx >= total) return;
    int xo = idx % Wout;
    int t  = idx / Wout;
    int yo = t % Hout;
    int n  = t / Hout;
    float mx = 0.0f;
    for (int ky = 0; ky < 3; ++ky) {
        int iy = 2 * yo + ky - 1;
        if (iy < 0 || iy >= Hin) continue;
        for (int kx = 0; kx < 3; ++kx) {
            int ix = 2 * xo + kx - 1;
            if (ix < 0 || ix >= Win) continue;
            mx = fmaxf(mx, (float)in[((size_t)n * Hin + iy) * Win + ix]);
        }
    }
    out[idx] = (_Float16)mx;
}

// ---------------------------------------------------------------------------
// inp[n,y,x,0..31] = { img*mask1 (3), mask*mask1 (1), zeros (28) }  NHWC f16
// ---------------------------------------------------------------------------
__global__ __launch_bounds__(256) void build_input_k(const float* __restrict__ image,
                                                     const float* __restrict__ masks,
                                                     const _Float16* __restrict__ m1,
                                                     _Float16* __restrict__ inp,
                                                     int N, int H, int W) {
    int idx = blockIdx.x * 256 + threadIdx.x;
    int total = N * H * W;
    if (idx >= total) return;
    int hw = H * W;
    int n  = idx / hw;
    int yx = idx - n * hw;
    int bb = n >> 2;  // n_i == 4
    float mv = (float)m1[idx];
    _Float16* o = inp + (size_t)idx * 32;
#pragma unroll
    for (int c = 0; c < 3; ++c)
        o[c] = (_Float16)(image[((size_t)bb * 3 + c) * hw + yx] * mv);
    o[3] = (_Float16)(masks[idx] * mv);
#pragma unroll
    for (int c = 4; c < 32; ++c) o[c] = (_Float16)0.0f;
}

// ---------------------------------------------------------------------------
// xs += broadcast(x)*mask4  (xs: NHWC f16 C=64, x: [2,64,96,96] f32 NCHW)
// ---------------------------------------------------------------------------
__global__ __launch_bounds__(256) void add_x_bcast_k(const float* __restrict__ x,
                                                     const _Float16* __restrict__ m4,
                                                     _Float16* __restrict__ xs,
                                                     int N, int H, int W) {
    int idx = blockIdx.x * 256 + threadIdx.x;
    int total = N * H * W * 64;
    if (idx >= total) return;
    int c   = idx & 63;
    int pix = idx >> 6;
    int hw  = H * W;
    int n   = pix / hw;
    int yx  = pix - n * hw;
    int bb  = n >> 2;
    float xv = x[((size_t)bb * 64 + c) * hw + yx] * (float)m4[pix];
    xs[idx] = (_Float16)((float)xs[idx] + xv);
}

// ---------------------------------------------------------------------------
// OIHW f32 -> [tap][Opad][Kpad] f16 (zero padded)
// ---------------------------------------------------------------------------
__global__ __launch_bounds__(256) void repack_w_k(const float* __restrict__ src,
                                                  _Float16* __restrict__ dst,
                                                  int O, int I, int Opad, int Kpad) {
    int idx = blockIdx.x * 256 + threadIdx.x;
    int total = 9 * Opad * Kpad;
    if (idx >= total) return;
    int k   = idx % Kpad;
    int o   = (idx / Kpad) % Opad;
    int tap = idx / (Kpad * Opad);
    float v = 0.0f;
    if (o < O && k < I) v = src[(size_t)(o * I + k) * 9 + tap];
    dst[((size_t)tap * Opad + o) * Kpad + k] = (_Float16)v;
}

// ---------------------------------------------------------------------------
// LDS-tiled implicit-GEMM 3x3 conv.
// Staging: TDM row DMAs (wave 0 issues, TENSORcnt) + zero halo; fallback is a
// cooperative vector copy. Compute: branchless unrolled WMMA chains from LDS.
// ---------------------------------------------------------------------------
template <int CIN, int STRIDE>
__global__ __launch_bounds__(256) void conv3x3_wmma_k(
    const _Float16* __restrict__ in, const _Float16* __restrict__ wgt,
    const _Float16* __restrict__ mask, const _Float16* __restrict__ res,
    _Float16* __restrict__ out, float* __restrict__ outf,
    const float* __restrict__ biasp,
    int Hin, int Win, int Hout, int Wout,
    int Cout, int mode, int xspan)
{
    extern __shared__ _Float16 smem[];

    const int tid     = threadIdx.x;
    const int xblocks = Wout / xspan;
    int bx = blockIdx.x;
    const int xblk = bx % xblocks;  bx /= xblocks;
    const int y    = bx % Hout;
    const int n    = bx / Hout;

    const int pixspan = xspan * STRIDE + 2;
    const int gx0     = xblk * xspan * STRIDE - 1;  // input-x of LDS column 0

#if ATHENA_TDM
    // ---- TDM staging: interior via tensor DMA, halo/OOB zero-filled ----
    const int c0 = (gx0 < 0) ? 1 : 0;                 // first in-range LDS column
    int c1 = Win - gx0; if (c1 > pixspan) c1 = pixspan; // one past last in-range col
    for (int r = 0; r < 3; ++r) {
        const int iy = y * STRIDE + r - 1;
        const bool rowOk = (iy >= 0) && (iy < Hin);
        if (!rowOk) {
            for (int c = tid; c < (pixspan * CIN) >> 3; c += 256)
                *(v8h*)(smem + (size_t)r * pixspan * CIN + (c << 3)) = zero_v8h();
        } else {
            if (c0 > 0)
                for (int c = tid; c < (CIN >> 3); c += 256)
                    *(v8h*)(smem + (size_t)r * pixspan * CIN + (c << 3)) = zero_v8h();
            if (c1 < pixspan)
                for (int c = tid; c < (CIN >> 3); c += 256)
                    *(v8h*)(smem + ((size_t)r * pixspan + (pixspan - 1)) * CIN + (c << 3)) = zero_v8h();
        }
    }
    if (tid < 32) {  // one wave issues the DMAs (TDM ignores EXEC within the wave)
        for (int r = 0; r < 3; ++r) {
            const int iy = y * STRIDE + r - 1;
            if (iy < 0 || iy >= Hin) continue;
            const _Float16* gsrc = in + (((size_t)n * Hin + iy) * Win + (gx0 + c0)) * CIN;
            _Float16* ldst = smem + ((size_t)r * pixspan + c0) * CIN;
            tdm_load_row(gsrc, ldst, (unsigned)((c1 - c0) * CIN) >> 1);  // dwords
        }
        __builtin_amdgcn_s_wait_tensorcnt(0);
    }
#else
    // ---- fallback cooperative stage: 3 rows x pixspan x CIN, halo zeroed ----
    const int count8 = (pixspan * CIN) >> 3;
    for (int c = tid; c < 3 * count8; c += 256) {
        const int r       = c / count8;
        const int o       = c - r * count8;
        const int halfoff = o << 3;
        const int pixel   = halfoff / CIN;
        const int chan    = halfoff & (CIN - 1);
        const int iy      = y * STRIDE + r - 1;
        const int ix      = gx0 + pixel;
        v8h vv;
        if (iy >= 0 && iy < Hin && ix >= 0 && ix < Win)
            vv = *(const v8h*)(in + (((size_t)n * Hin + iy) * Win + ix) * CIN + chan);
        else
            vv = zero_v8h();
        *(v8h*)(smem + ((size_t)r * pixspan + pixel) * CIN + chan) = vv;
    }
#endif
    __syncthreads();

    // ---- per-wave tiles: 16 pixels x 16 out channels ----
    const int wave = tid >> 5;
    const int lane = tid & 31;
    const int hi   = lane >> 4;   // K-half selector (this code's contiguous k = 16*hi + s order)
    const int c15  = lane & 15;   // A row (pixel) / B column (out channel)

    const int xsubs   = xspan >> 4;
    const int cgroups = Cout >> 4;
    const int tiles   = xsubs * cgroups;
    const float bias  = (biasp != nullptr) ? biasp[0] : 0.0f;

    for (int t = wave; t < tiles; t += 8) {
        const int cg = t / xsubs;
        const int xs = t - cg * xsubs;

        v8f acc;
#pragma unroll
        for (int r = 0; r < 8; ++r) acc[r] = 0.0f;

#pragma unroll
        for (int ky = 0; ky < 3; ++ky) {
#pragma unroll
            for (int kx = 0; kx < 3; ++kx) {
                const int lix = (xs * 16 + c15) * STRIDE + kx;   // local input x
                const _Float16* ab   = smem + ((size_t)ky * pixspan + lix) * CIN;
                const _Float16* wrow =
                    wgt + ((size_t)(ky * 3 + kx) * Cout + (cg * 16 + c15)) * CIN;
#pragma unroll
                for (int kb = 0; kb < CIN; kb += 32) {
                    v16h a = *(const v16h*)(ab + kb + hi * 16);     // K = hi*16..+15
                    v16h bfrag = *(const v16h*)(wrow + kb + hi * 16); // K = hi*16..+15
                    acc = __builtin_amdgcn_wmma_f32_16x16x32_f16(
                              false, a, false, bfrag, (short)0, acc, false, false);
                }
            }
        }

        const int oc = cg * 16 + c15;
#pragma unroll
        for (int r = 0; r < 8; ++r) {
            const int    m   = r + (hi << 3);   // D: lanes 0-15 -> M=r, 16-31 -> M=8+r
            const int    xo  = xblk * xspan + xs * 16 + m;
            const size_t pix = ((size_t)n * Hout + y) * Wout + xo;
            float v  = acc[r];
            const float mk = (float)mask[pix];
            if (mode == EP_RELU_BN) {
                v = fmaxf(v * mk, 0.0f) * BN_S;
                out[pix * Cout + oc] = (_Float16)v;
            } else if (mode == EP_LEAKY_BN) {
                v = v * mk * BN_S;
                v = (v > 0.0f) ? v : 0.2f * v;
                out[pix * Cout + oc] = (_Float16)v;
            } else if (mode == EP_ADD) {
                v = v * mk + (float)res[pix * Cout + oc];
                out[pix * Cout + oc] = (_Float16)v;
            } else { // EP_WHERE: only real channel 0 exists
                if (oc == 0) outf[pix] = (mk > 0.0f) ? (v + bias) : -99.0f;
            }
        }
    }
}

// ---------------------------------------------------------------------------
// LDS-tiled transposed conv (SparseInverseConv adjoint of stride-2/pad-1/k=3).
// Even outputs use only the center tap per dim; odd outputs use taps {0,2}.
// Block stages 2 coarse rows x (xspan/2+1) cols; each wave computes 16
// SAME-PARITY pixels x 16 out-channels with uniform taps, A from LDS.
// Epilogue: leaky(v*mask*BN_S).
// ---------------------------------------------------------------------------
template <int CIN>
__global__ __launch_bounds__(256) void tconv3x3_wmma_k(
    const _Float16* __restrict__ in, const _Float16* __restrict__ wgt,
    const _Float16* __restrict__ mask, _Float16* __restrict__ out,
    int Hin, int Win, int Hout, int Wout, int Cout, int xspan)
{
    extern __shared__ _Float16 smem[];
    const int tid = threadIdx.x;
    const int xblocks = Wout / xspan;
    int bx = blockIdx.x;
    const int xblk = bx % xblocks;  bx /= xblocks;
    const int y    = bx % Hout;
    const int n    = bx / Hout;

    const int cols = (xspan >> 1) + 1;   // coarse columns needed
    const int cx0  = (xblk * xspan) >> 1;
    const int iy0  = y >> 1;

    // ---- stage 2 coarse rows (zero-filled OOB) ----
    const int count8 = (cols * CIN) >> 3;
    for (int c = tid; c < 2 * count8; c += 256) {
        const int r       = c / count8;
        const int o       = c - r * count8;
        const int halfoff = o << 3;
        const int col     = halfoff / CIN;
        const int ch      = halfoff & (CIN - 1);
        const int iy      = iy0 + r;
        const int ix      = cx0 + col;
        v8h vv;
        if (iy < Hin && ix < Win)
            vv = *(const v8h*)(in + (((size_t)n * Hin + iy) * Win + ix) * CIN + ch);
        else
            vv = zero_v8h();
        *(v8h*)(smem + ((size_t)r * cols + col) * CIN + ch) = vv;
    }
    __syncthreads();

    const int wave = tid >> 5;
    const int lane = tid & 31;
    const int hi   = lane >> 4;
    const int c15  = lane & 15;

    const int xsubs   = xspan >> 5;       // 32-column groups (16 px per parity)
    const int cgroups = Cout >> 4;
    const int tiles   = xsubs * 2 * cgroups;

    for (int t = wave; t < tiles; t += 8) {
        int tt = t;
        const int cg  = tt % cgroups; tt /= cgroups;
        const int par = tt & 1;       tt >>= 1;
        const int xs  = tt;

        v8f acc;
#pragma unroll
        for (int r = 0; r < 8; ++r) acc[r] = 0.0f;

#pragma unroll
        for (int ky = 0; ky < 3; ++ky) {
            const int ty = y + ky - 1;
            if (ty & 1) continue;                 // uniform: odd dilated row -> 0
            const int lr = (ty >> 1) - iy0;       // 0 or 1
#pragma unroll
            for (int kx = 0; kx < 3; ++kx) {
                if ((par + kx - 1) & 1) continue; // uniform parity mismatch
                const int off = (par + kx - 1) >> 1;          // 0 or 1
                const int col = xs * 16 + c15 + off;
                const _Float16* ab  = smem + ((size_t)lr * cols + col) * CIN;
                const int tap = (2 - ky) * 3 + (2 - kx);      // flipped kernel
                const _Float16* wrow =
                    wgt + ((size_t)tap * Cout + (cg * 16 + c15)) * CIN;
#pragma unroll
                for (int kb = 0; kb < CIN; kb += 32) {
                    v16h a = *(const v16h*)(ab + kb + hi * 16);
                    v16h bfrag = *(const v16h*)(wrow + kb + hi * 16);
                    acc = __builtin_amdgcn_wmma_f32_16x16x32_f16(
                              false, a, false, bfrag, (short)0, acc, false, false);
                }
            }
        }

        const int oc = cg * 16 + c15;
#pragma unroll
        for (int r = 0; r < 8; ++r) {
            const int    m   = r + (hi << 3);
            const int    xo  = xblk * xspan + xs * 32 + 2 * m + par;
            const size_t pix = ((size_t)n * Hout + y) * Wout + xo;
            float v  = acc[r];
            const float mk = (float)mask[pix];
            v = v * mk * BN_S;
            v = (v > 0.0f) ? v : 0.2f * v;
            out[pix * Cout + oc] = (_Float16)v;
        }
    }
}

// ---------------------------------------------------------------------------
static inline int cdiv_i(int a, int b) { return (a + b - 1) / b; }

extern "C" void kernel_launch(void* const* d_in, const int* in_sizes, int n_in,
                              void* d_out, int out_size, void* d_ws, size_t ws_size,
                              hipStream_t stream) {
    (void)in_sizes; (void)n_in; (void)out_size; (void)ws_size;
    const int N = 8, H = 384, W = 384, H2 = 192, W2 = 192, H4 = 96, W4 = 96;

    const float* x_in  = (const float*)d_in[0];
    const float* image = (const float*)d_in[1];
    const float* masks = (const float*)d_in[2];
    const float* roi   = (const float*)d_in[3];
    const float* r4_b2 = (const float*)d_in[16];
    const float* r1_b2 = (const float*)d_in[19];

    float* o4 = (float*)d_out;             // [N,96,96]
    float* o1 = o4 + (size_t)N * H4 * W4;  // [N,384,384]

    // ---- carve workspace (f16 halves) ----
    char* p = (char*)d_ws;
    auto take = [&](size_t halves) -> _Float16* {
        _Float16* r = (_Float16*)p;
        p += (halves * sizeof(_Float16) + 255) & ~(size_t)255;
        return r;
    };

    // repacked weights: {input idx, O, I, Opad, Kpad}
    struct WDesc { int src, O, I, Op, Kp; };
    const WDesc wl[14] = {
        {4, 32, 4, 32, 32},   // w1a (Cin padded 4->32)
        {5, 32, 32, 32, 32},  // w1b
        {6, 32, 32, 32, 32},  // w2a
        {7, 32, 32, 32, 32},  // w2b
        {8, 64, 32, 64, 32},  // w4a
        {9, 64, 64, 64, 64},  // w4b
        {10, 64, 64, 64, 64}, // inv4_w
        {11, 32, 64, 32, 64}, // l4_sub_w
        {12, 32, 32, 32, 32}, // inv2_w
        {13, 32, 32, 32, 32}, // l5_sub_w
        {14, 32, 64, 32, 64}, // r4_w1
        {15, 1, 32, 16, 32},  // r4_w2 (O padded 1->16)
        {17, 32, 32, 32, 32}, // r1_w1
        {18, 1, 32, 16, 32},  // r1_w2
    };
    _Float16* Wp[14];
    for (int i = 0; i < 14; ++i) Wp[i] = take((size_t)9 * wl[i].Op * wl[i].Kp);

    _Float16* M1 = take((size_t)N * H * W);
    _Float16* M2 = take((size_t)N * H2 * W2);
    _Float16* M4 = take((size_t)N * H4 * W4);
    _Float16* B0 = take((size_t)N * H * W * 32);   // inp / v / t1
    _Float16* B1 = take((size_t)N * H * W * 32);   // f / v2
    _Float16* B2 = take((size_t)N * H * W * 32);   // fea1 (live until l5 add)
    _Float16* C0 = take((size_t)N * H2 * W2 * 32); // g / u2
    _Float16* C1 = take((size_t)N * H2 * W2 * 32); // fea2 (live until l4 add)
    _Float16* C2 = take((size_t)N * H2 * W2 * 64); // u
    _Float16* D0 = take((size_t)N * H4 * W4 * 64); // h / t
    _Float16* D1 = take((size_t)N * H4 * W4 * 64); // fea3 / xs

    // ---- weight repack ----
    for (int i = 0; i < 14; ++i) {
        int tot = 9 * wl[i].Op * wl[i].Kp;
        repack_w_k<<<cdiv_i(tot, 256), 256, 0, stream>>>(
            (const float*)d_in[wl[i].src], Wp[i], wl[i].O, wl[i].I, wl[i].Op, wl[i].Kp);
    }

    // ---- masks + packed input ----
    build_mask1_k<<<cdiv_i(N * H * W, 256), 256, 0, stream>>>(roi, M1, N * H * W);
    pool_mask_k<<<cdiv_i(N * H2 * W2, 256), 256, 0, stream>>>(M1, M2, N, H, W, H2, W2);
    pool_mask_k<<<cdiv_i(N * H4 * W4, 256), 256, 0, stream>>>(M2, M4, N, H2, W2, H4, W4);
    build_input_k<<<cdiv_i(N * H * W, 256), 256, 0, stream>>>(image, masks, M1, B0, N, H, W);

    // xspan per call: Wout % xspan == 0, LDS = 3*(xspan*stride+2)*Cin*2 <= ~25KB
    auto conv = [&](const _Float16* in, const _Float16* wg, const _Float16* mk,
                    const _Float16* res, _Float16* out, float* outf, const float* bp,
                    int Hin, int Win, int Hout, int Wout, int Cin, int Cout,
                    int stride, int mode, int xspan) {
        int blocks  = N * Hout * (Wout / xspan);
        int pixspan = xspan * stride + 2;
        size_t lds  = (size_t)3 * pixspan * Cin * sizeof(_Float16);
        if (Cin == 32 && stride == 1)
            conv3x3_wmma_k<32, 1><<<blocks, 256, lds, stream>>>(
                in, wg, mk, res, out, outf, bp, Hin, Win, Hout, Wout, Cout, mode, xspan);
        else if (Cin == 32 && stride == 2)
            conv3x3_wmma_k<32, 2><<<blocks, 256, lds, stream>>>(
                in, wg, mk, res, out, outf, bp, Hin, Win, Hout, Wout, Cout, mode, xspan);
        else
            conv3x3_wmma_k<64, 1><<<blocks, 256, lds, stream>>>(
                in, wg, mk, res, out, outf, bp, Hin, Win, Hout, Wout, Cout, mode, xspan);
    };
    auto tconv = [&](const _Float16* in, const _Float16* wg, const _Float16* mk,
                     _Float16* out, int Hin, int Win, int Hout, int Wout,
                     int Cin, int Cout, int xspan) {
        int blocks = N * Hout * (Wout / xspan);
        size_t lds = (size_t)2 * ((xspan >> 1) + 1) * Cin * sizeof(_Float16);
        if (Cin == 32)
            tconv3x3_wmma_k<32><<<blocks, 256, lds, stream>>>(
                in, wg, mk, out, Hin, Win, Hout, Wout, Cout, xspan);
        else
            tconv3x3_wmma_k<64><<<blocks, 256, lds, stream>>>(
                in, wg, mk, out, Hin, Win, Hout, Wout, Cout, xspan);
    };

    // ---- encoder ----
    conv(B0, Wp[0], M1, nullptr, B1, nullptr, nullptr, H, W, H, W, 32, 32, 1, EP_RELU_BN, 64);    // f
    conv(B1, Wp[1], M1, nullptr, B2, nullptr, nullptr, H, W, H, W, 32, 32, 1, EP_RELU_BN, 64);    // fea1
    conv(B2, Wp[2], M2, nullptr, C0, nullptr, nullptr, H, W, H2, W2, 32, 32, 2, EP_RELU_BN, 64);  // g
    conv(C0, Wp[3], M2, nullptr, C1, nullptr, nullptr, H2, W2, H2, W2, 32, 32, 1, EP_RELU_BN, 64);// fea2
    conv(C1, Wp[4], M4, nullptr, D0, nullptr, nullptr, H2, W2, H4, W4, 32, 64, 2, EP_RELU_BN, 32);// h
    conv(D0, Wp[5], M4, nullptr, D1, nullptr, nullptr, H4, W4, H4, W4, 64, 64, 1, EP_RELU_BN, 32);// fea3

    // xs = fea3 + broadcast(x)*mask4   (in place in D1)
    add_x_bcast_k<<<cdiv_i(N * H4 * W4 * 64, 256), 256, 0, stream>>>(x_in, M4, D1, N, H4, W4);

    // ---- refine_OS4 ----
    conv(D1, Wp[10], M4, nullptr, D0, nullptr, nullptr, H4, W4, H4, W4, 64, 32, 1, EP_LEAKY_BN, 48); // t
    conv(D0, Wp[11], M4, nullptr, nullptr, o4, r4_b2, H4, W4, H4, W4, 32, 16, 1, EP_WHERE, 96);      // x_os4_out

    // ---- decoder: layer4 -> layer5 ----
    tconv(D1, Wp[6], M2, C2, H4, W4, H2, W2, 64, 64, 64);                                            // u (leaky)
    conv(C2, Wp[7], M2, C1, C0, nullptr, nullptr, H2, W2, H2, W2, 64, 32, 1, EP_ADD, 64);            // u2 = conv+fea2
    tconv(C0, Wp[8], M1, B0, H2, W2, H, W, 32, 32, 64);                                              // v (leaky)
    conv(B0, Wp[9], M1, B2, B1, nullptr, nullptr, H, W, H, W, 32, 32, 1, EP_ADD, 64);                // v2 = conv+fea1

    // ---- refine_OS1 ----
    conv(B1, Wp[12], M1, nullptr, B0, nullptr, nullptr, H, W, H, W, 32, 32, 1, EP_LEAKY_BN, 64);     // t1
    conv(B0, Wp[13], M1, nullptr, nullptr, o1, r1_b2, H, W, H, W, 32, 16, 1, EP_WHERE, 128);         // x_os1_out
}
